// Model_76974403879582
// MI455X (gfx1250) — 8-bit matrix operand order repaired and checked by reading, compiled, not run on hardware
//
#include <hip/hip_runtime.h>
#include <hip/hip_bf16.h>

typedef __attribute__((ext_vector_type(16))) int   v16i;
typedef __attribute__((ext_vector_type(8)))  float v8f;

constexpr int M_TOT  = 8192;
constexpr int N_TOT  = 4096;
constexpr int K_TOT  = 4096;
constexpr int GROUPS = 8;
constexpr int KB     = K_TOT / 128;
constexpr int NBLK   = N_TOT / 128;

#if defined(__has_builtin)
#  if __has_builtin(__builtin_amdgcn_cvt_pk_fp8_f32)
#    define USE_HW_FP8_CVT 1
#  endif
#endif
#ifndef USE_HW_FP8_CVT
#  define USE_HW_FP8_CVT 0
#endif

__device__ inline unsigned f32_to_e4m3(float x) {
  unsigned u   = __float_as_uint(x);
  unsigned s   = (u >> 24) & 0x80u;
  unsigned ab  = u & 0x7FFFFFFFu;
  if (ab >= 0x43E00000u) return s | 0x7Eu;
  if (ab < 0x3C800000u) {
    unsigned q = (unsigned)__float2int_rn(__uint_as_float(ab) * 512.0f);
    if (q >= 8u) return s | 0x08u;
    return s | q;
  }
  unsigned e   = ab >> 23;
  unsigned man = ab & 0x7FFFFFu;
  unsigned m3  = man >> 20;
  unsigned rem = man & 0xFFFFFu;
  m3 += (rem > 0x80000u) || ((rem == 0x80000u) && (m3 & 1u));
  unsigned v = ((e - 120u) << 3) + m3;
  if (v > 0x7Eu) v = 0x7Eu;
  return s | v;
}

__device__ inline unsigned pack_fp8x4(float x0, float x1, float x2, float x3) {
#if USE_HW_FP8_CVT
  int p = __builtin_amdgcn_cvt_pk_fp8_f32(x0, x1, 0, false);
  p     = __builtin_amdgcn_cvt_pk_fp8_f32(x2, x3, p, true);
  return (unsigned)p;
#else
  return f32_to_e4m3(x0) | (f32_to_e4m3(x1) << 8) |
         (f32_to_e4m3(x2) << 16) | (f32_to_e4m3(x3) << 24);
#endif
}

__global__ __launch_bounds__(256) void fp8_group_gemm(
    const float* __restrict__ a,
    const float* __restrict__ a_sf,
    const float* __restrict__ b,
    const float* __restrict__ b_sf,
    const int*   __restrict__ midx,
    __hip_bfloat16* __restrict__ out)
{
  __shared__ __align__(64) int   As[2][8][32][16];
  __shared__ __align__(64) int   Bs[2][8][32][16];
  __shared__ float aS[2][128];
  __shared__ unsigned gmaskSh;

  const int tid     = threadIdx.x;
  const int rowBase = blockIdx.y * 128;
  const int colBase = blockIdx.x * 128;
  const int nTile   = blockIdx.x;

  if (tid == 0) gmaskSh = 0u;
  __syncthreads();
  if (tid < 128) {
    int gi = midx[rowBase + tid];
    if (gi >= 0) atomicOr(&gmaskSh, 1u << gi);
  }
  __syncthreads();
  const unsigned gmask = gmaskSh;

  const int lsub  = tid >> 5;
  const int llane = tid & 31;
  const int llo   = llane & 15;
  const int lhi   = llane >> 4;
  const int aRow  = rowBase + lsub * 16 + llo;
  const int myG   = midx[aRow];
  const int bCol  = colBase + lsub * 16 + llo;

  const int wid    = tid >> 5;
  const int lane   = tid & 31;
  const int wm     = wid >> 1;
  const int wn     = wid & 1;
  const int laneHi = lane >> 4;
  const int laneLo = lane & 15;

  float acc[2][4][8];
#pragma unroll
  for (int ms = 0; ms < 2; ++ms)
#pragma unroll
    for (int ns = 0; ns < 4; ++ns)
#pragma unroll
      for (int vg = 0; vg < 8; ++vg) acc[ms][ns][vg] = 0.0f;

  auto load_tile = [&](int g, int ks, int buf) {
    const float* Ar  = a + (size_t)aRow * K_TOT + (size_t)ks * 128;
    const bool   act = (myG == g);
#pragma unroll
    for (int gq = 0; gq < 4; ++gq) {
      int kb = 64 * (gq >> 1) + 32 * (gq & 1) + 8 * lhi;
      int4 pk;
      if (act) {
        float4 f0 = *(const float4*)(Ar + kb + 0);
        float4 f1 = *(const float4*)(Ar + kb + 4);
        float4 f2 = *(const float4*)(Ar + kb + 16);
        float4 f3 = *(const float4*)(Ar + kb + 20);
        pk.x = (int)pack_fp8x4(f0.x, f0.y, f0.z, f0.w);
        pk.y = (int)pack_fp8x4(f1.x, f1.y, f1.z, f1.w);
        pk.z = (int)pack_fp8x4(f2.x, f2.y, f2.z, f2.w);
        pk.w = (int)pack_fp8x4(f3.x, f3.y, f3.z, f3.w);
      } else {
        pk = make_int4(0, 0, 0, 0);
      }
      *(int4*)&As[buf][lsub][llane][gq * 4] = pk;
    }
    const float* Br = b + ((size_t)g * N_TOT + bCol) * K_TOT + (size_t)ks * 128;
#pragma unroll
    for (int gq = 0; gq < 4; ++gq) {
      int kb = 32 * gq + 8 * lhi;
      float4 f0 = *(const float4*)(Br + kb + 0);
      float4 f1 = *(const float4*)(Br + kb + 4);
      float4 f2 = *(const float4*)(Br + kb + 16);
      float4 f3 = *(const float4*)(Br + kb + 20);
      int4 pk;
      pk.x = (int)pack_fp8x4(f0.x, f0.y, f0.z, f0.w);
      pk.y = (int)pack_fp8x4(f1.x, f1.y, f1.z, f1.w);
      pk.z = (int)pack_fp8x4(f2.x, f2.y, f2.z, f2.w);
      pk.w = (int)pack_fp8x4(f3.x, f3.y, f3.z, f3.w);
      *(int4*)&Bs[buf][lsub][llane][gq * 4] = pk;
    }
    if (tid < 128) aS[buf][tid] = a_sf[(size_t)(rowBase + tid) * KB + ks];
  };

  for (int g = 0; g < GROUPS; ++g) {
    if (!(gmask & (1u << g))) continue;

    load_tile(g, 0, 0);
    __syncthreads();

    for (int ks = 0; ks < KB; ++ks) {
      const int cur = ks & 1;
      if (ks + 1 < KB) load_tile(g, ks + 1, cur ^ 1);

      const float sB = b_sf[((size_t)g * NBLK + nTile) * KB + ks];

      float sA[2][8];
#pragma unroll
      for (int ms = 0; ms < 2; ++ms)
#pragma unroll
        for (int vg = 0; vg < 8; ++vg)
          sA[ms][vg] = aS[cur][wm * 32 + ms * 16 + vg + 8 * laneHi] * sB;

      const v16i afr0 = *(const v16i*)&As[cur][wm * 2 + 0][lane][0];
      const v16i afr1 = *(const v16i*)&As[cur][wm * 2 + 1][lane][0];
      const v8f  zc   = {};

#pragma unroll
      for (int ns = 0; ns < 4; ++ns) {
        const v16i bfr = *(const v16i*)&Bs[cur][wn * 4 + ns][lane][0];
        v8f d0 = __builtin_amdgcn_wmma_f32_16x16x128_fp8_fp8(afr0, bfr, (short)0, zc, false, false);
#pragma unroll
        for (int vg = 0; vg < 8; ++vg)
          acc[0][ns][vg] = fmaf(sA[0][vg], d0[vg], acc[0][ns][vg]);
        v8f d1 = __builtin_amdgcn_wmma_f32_16x16x128_fp8_fp8(afr1, bfr, (short)0, zc, false, false);
#pragma unroll
        for (int vg = 0; vg < 8; ++vg)
          acc[1][ns][vg] = fmaf(sA[1][vg], d1[vg], acc[1][ns][vg]);
      }
      __syncthreads();
    }
  }

#pragma unroll
  for (int ms = 0; ms < 2; ++ms) {
    const int r0 = rowBase + wm * 32 + ms * 16 + 8 * laneHi;
#pragma unroll
    for (int ns = 0; ns < 4; ++ns) {
      const int c = colBase + wn * 64 + ns * 16 + laneLo;
#pragma unroll
      for (int vg = 0; vg < 8; ++vg)
        out[(size_t)(r0 + vg) * N_TOT + c] = __float2bfloat16(acc[ms][ns][vg]);
    }
  }
}

extern "C" void kernel_launch(void* const* d_in, const int* in_sizes, int n_in,
                              void* d_out, int out_size, void* d_ws, size_t ws_size,
                              hipStream_t stream) {
  (void)in_sizes; (void)n_in; (void)out_size; (void)d_ws; (void)ws_size;
  const float* a    = (const float*)d_in[0];
  const float* a_sf = (const float*)d_in[1];
  const float* b    = (const float*)d_in[2];
  const float* b_sf = (const float*)d_in[3];
  const int*   midx = (const int*)d_in[4];
  __hip_bfloat16* out = (__hip_bfloat16*)d_out;

  dim3 grid(N_TOT / 128, M_TOT / 128);
  fp8_group_gemm<<<grid, 256, 0, stream>>>(a, a_sf, b, b_sf, midx, out);
}
